// SpearmanCorrelation_46291157516693
// MI455X (gfx1250) — matrix operand order repaired and checked by reading, not compiled, not run on hardware
//
#include <hip/hip_runtime.h>

typedef __attribute__((ext_vector_type(16))) _Float16 v16h;
typedef __attribute__((ext_vector_type(8)))  _Float16 v8h;
typedef __attribute__((ext_vector_type(8)))  float    v8f;

#define E_DIM 128
#define F_DIM 32
#define LDW   136   // padded LDS row stride in halves (272B -> 4-bank rotate/row)

__global__ void zero_f32_kernel(float* __restrict__ p, int n) {
    int i = blockIdx.x * blockDim.x + threadIdx.x;
    if (i < n) p[i] = 0.0f;
}

__global__ __launch_bounds__(256)
void spearman_corr_kernel(const float* __restrict__ de, float* __restrict__ out) {
    const int b    = blockIdx.x;      // batch
    const int f    = blockIdx.y;      // freq band
    const int tid  = threadIdx.x;
    const int lane = tid & 31;
    const int wave = tid >> 5;        // 0..7

    __shared__ float xs[E_DIM];
    __shared__ float rs[E_DIM];
    __shared__ float se[E_DIM];
    __shared__ alignas(16) _Float16 W[E_DIM * LDW];

    // ---- Stage 1: load de_data[b, :, f] (stride F) ----
    if (tid < E_DIM) {
        xs[tid] = de[((size_t)b * E_DIM + tid) * F_DIM + f];
    }
    __syncthreads();

    // ---- Stage 2: ranks = argsort(argsort(x)); ties broken by index ----
    if (tid < E_DIM) {
        float xi = xs[tid];
        int r = 0;
#pragma unroll 8
        for (int j = 0; j < E_DIM; ++j) {
            float xj = xs[j];
            r += (int)((xj < xi) || (xj == xi && j < tid));
        }
        rs[tid] = (float)r;
    }
    __syncthreads();

    // ---- Stage 3: W[j][k] = 1/(|r_j - r_k| + 1) as f16; se = col sums (f32) ----
    for (int idx = tid; idx < E_DIM * E_DIM; idx += 256) {
        int row = idx >> 7;
        int col = idx & (E_DIM - 1);
        float w = 1.0f / (fabsf(rs[row] - rs[col]) + 1.0f);
        W[row * LDW + col] = (_Float16)w;
    }
    if (tid < E_DIM) {
        float rk = rs[tid];
        float s = 0.0f;
#pragma unroll 8
        for (int j = 0; j < E_DIM; ++j)
            s += 1.0f / (fabsf(rs[j] - rk) + 1.0f);
        se[tid] = s;
    }
    __syncthreads();

    // ---- Stage 4: gram = W * W via WMMA (W symmetric => B rows == W rows) ----
    const int m_tile = wave * 16;          // this wave's 16-row band
    const int kgrp   = lane >> 4;          // 0 or 1 (lane half)
    const int l16    = lane & 15;
    const int row_m  = m_tile + l16;       // A-fragment row for this lane

    const float inv_e1 = 1.0f / (float)(E_DIM - 1);
    const float inv_f  = 1.0f / (float)F_DIM;

    // Hoist: A fragments are n_tile-invariant -> load all 4 K-steps once.
    // A (16x32 f16) layout: VGPR0-3 -> K = kgrp*8 + 0..7, VGPR4-7 -> +16
    v16h afrag[4];
#pragma unroll
    for (int ks = 0; ks < 4; ++ks) {
        const int k0 = ks * 32;
        const v8h a0 = *(const v8h*)&W[row_m * LDW + k0 + kgrp * 8];
        const v8h a1 = *(const v8h*)&W[row_m * LDW + k0 + kgrp * 8 + 16];
        afrag[ks] = __builtin_shufflevector(a0, a1,
            0, 1, 2, 3, 4, 5, 6, 7, 8, 9, 10, 11, 12, 13, 14, 15);
    }

    // Hoist: se[j]*inv_e1*inv_f for this wave's 8 output rows (per C/D layout:
    // VGPR r holds M = m_tile + kgrp*8 + r).
    float sej[8];
#pragma unroll
    for (int r = 0; r < 8; ++r) {
        sej[r] = se[m_tile + kgrp * 8 + r] * inv_e1 * inv_f;
    }

    for (int n_tile = 0; n_tile < E_DIM; n_tile += 16) {
        const int row_n = n_tile + l16;    // B-fragment row (symmetry)
        v8f c = {};
#pragma unroll
        for (int ks = 0; ks < 4; ++ks) {
            const int k0 = ks * 32;
            // B (32x16 f16): VGPR0-3 -> K = kgrp*8 + 0..7, VGPR4-7 -> +16 (matches A)
            const v8h b0 = *(const v8h*)&W[row_n * LDW + k0 + kgrp * 8];
            const v8h b1 = *(const v8h*)&W[row_n * LDW + k0 + kgrp * 8 + 16];
            v16h bb = __builtin_shufflevector(b0, b1,
                0, 1, 2, 3, 4, 5, 6, 7, 8, 9, 10, 11, 12, 13, 14, 15);

            c = __builtin_amdgcn_wmma_f32_16x16x32_f16(
                    /*neg_a=*/false, afrag[ks], /*neg_b=*/false, bb,
                    /*c_mod=*/(short)0, c, /*reuse_a=*/false, /*reuse_b=*/false);
        }

        // Epilogue: corr = gram - se_j*se_k/(E-1); accumulate mean over F
        const float sk = se[n_tile + l16];
        float* dst_base = &out[((size_t)b * E_DIM + m_tile + kgrp * 8) * E_DIM
                               + n_tile + l16];
#pragma unroll
        for (int r = 0; r < 8; ++r) {
            const float corr = c[r] * inv_f - sej[r] * sk;
            unsafeAtomicAdd(dst_base + (size_t)r * E_DIM, corr);
        }
    }
}

extern "C" void kernel_launch(void* const* d_in, const int* in_sizes, int n_in,
                              void* d_out, int out_size, void* d_ws, size_t ws_size,
                              hipStream_t stream) {
    const float* de  = (const float*)d_in[0];
    float*       out = (float*)d_out;

    const int B = in_sizes[0] / (E_DIM * F_DIM);   // 128

    // Deterministic accumulation target: zero the output every launch.
    zero_f32_kernel<<<(out_size + 255) / 256, 256, 0, stream>>>(out, out_size);

    dim3 grid(B, F_DIM);
    spearman_corr_kernel<<<grid, 256, 0, stream>>>(de, out);
}
